// TransformerBlock_17008070492993
// MI455X (gfx1250) — matrix operand order repaired and checked by reading, not compiled, not run on hardware
//
#include <hip/hip_runtime.h>

// ---------------- CDNA5 types ----------------
typedef __attribute__((ext_vector_type(16))) __bf16 v16bf;
typedef __attribute__((ext_vector_type(8)))  __bf16 v8bf;
typedef __attribute__((ext_vector_type(8)))  float  v8f;
typedef __attribute__((ext_vector_type(4)))  int    v4i_;

// ---------------- problem constants ----------------
constexpr int Bc   = 2;
constexpr int Sc   = 2048;
constexpr int Dc   = 1024;
constexpr int Hc   = 16;
constexpr int HDc  = 64;      // head dim
constexpr int HIDc = 3584;
constexpr int Mc   = Bc * Sc; // 4096 rows

// ---------------- CDNA5 async global->LDS (guarded) ----------------
#if defined(__AMDGCN__) && __has_builtin(__builtin_amdgcn_global_load_async_to_lds_b128) && \
    __has_builtin(__builtin_amdgcn_s_wait_asynccnt)
#define USE_ASYNC_LDS 1
#else
#define USE_ASYNC_LDS 0
#endif

#if USE_ASYNC_LDS
__device__ __forceinline__ void async_cp16(const void* g, void* l) {
  __builtin_amdgcn_global_load_async_to_lds_b128(
      (__attribute__((address_space(1))) v4i_*)(uintptr_t)g,
      (__attribute__((address_space(3))) v4i_*)(unsigned int)(uintptr_t)l,
      0, 0);
}
__device__ __forceinline__ void async_wait0() { __builtin_amdgcn_s_wait_asynccnt(0); }
#else
__device__ __forceinline__ void async_wait0() {}
#endif

// ---------------- helpers ----------------
__device__ __forceinline__ unsigned short f2b(float f) {
  union { float f; unsigned u; } c; c.f = f;
  unsigned u = c.u;
  u += 0x7FFFu + ((u >> 16) & 1u);     // round-to-nearest-even
  return (unsigned short)(u >> 16);
}

__device__ __forceinline__ v16bf cat16(v8bf lo, v8bf hi) {
  v16bf r;
#pragma unroll
  for (int i = 0; i < 8; i++) { r[i] = lo[i]; r[i + 8] = hi[i]; }
  return r;
}

// DPP cross-lane permute within 16-lane rows (VALU, no LDS traffic)
template <int CTRL>
__device__ __forceinline__ float dpp_perm(float x) {
  return __int_as_float(
      __builtin_amdgcn_update_dpp(0, __float_as_int(x), CTRL, 0xF, 0xF, true));
}
// full 16-lane reductions: xor1, xor2, then cross-4 (row_half_mirror), cross-8 (row_mirror)
__device__ __forceinline__ float redmax16(float v) {
  v = fmaxf(v, dpp_perm<0xB1>(v));    // quad_perm [1,0,3,2]
  v = fmaxf(v, dpp_perm<0x4E>(v));    // quad_perm [2,3,0,1]
  v = fmaxf(v, dpp_perm<0x141>(v));   // row_half_mirror
  v = fmaxf(v, dpp_perm<0x140>(v));   // row_mirror
  return v;
}
__device__ __forceinline__ float redsum16(float v) {
  v += dpp_perm<0xB1>(v);
  v += dpp_perm<0x4E>(v);
  v += dpp_perm<0x141>(v);
  v += dpp_perm<0x140>(v);
  return v;
}

// ---------------- fp32 -> bf16 convert (weights) ----------------
__global__ void k_f32_to_bf16(const float* __restrict__ s,
                              unsigned short* __restrict__ d, size_t n) {
  for (size_t i = (size_t)blockIdx.x * blockDim.x + threadIdx.x; i < n;
       i += (size_t)gridDim.x * blockDim.x)
    d[i] = f2b(s[i]);
}

// ---------------- (x + pos) + RMSNorm -> bf16 ----------------
template <bool ADDPOS>
__global__ __launch_bounds__(256) void k_rmsnorm(
    const float* __restrict__ xin, const float* __restrict__ pos,
    const float* __restrict__ w, float* __restrict__ xr_out,
    unsigned short* __restrict__ hout) {
  const int row = blockIdx.x;
  const int tid = threadIdx.x;
  const float* xp = xin + (size_t)row * Dc;
  float4 v = *(const float4*)(xp + tid * 4);
  if (ADDPOS) {
    const float* pp = pos + (size_t)(row % Sc) * Dc;
    float4 p = *(const float4*)(pp + tid * 4);
    v.x += p.x; v.y += p.y; v.z += p.z; v.w += p.w;
    *(float4*)(xr_out + (size_t)row * Dc + tid * 4) = v;
  }
  float ss = v.x * v.x + v.y * v.y + v.z * v.z + v.w * v.w;
  ss = redsum16(ss);
  ss += __shfl_xor(ss, 16, 32);  // cross-half
  __shared__ float wsum[8];
  __shared__ float total;
  if ((tid & 31) == 0) wsum[tid >> 5] = ss;
  __syncthreads();
  if (tid == 0) {
    float t = 0.f;
#pragma unroll
    for (int i = 0; i < 8; i++) t += wsum[i];
    total = t;
  }
  __syncthreads();
  const float rstd = rsqrtf(total / (float)Dc + 1e-6f);
  float4 wv = *(const float4*)(w + tid * 4);
  unsigned short* op = hout + (size_t)row * Dc + tid * 4;
  op[0] = f2b(v.x * rstd * wv.x);
  op[1] = f2b(v.y * rstd * wv.y);
  op[2] = f2b(v.z * rstd * wv.z);
  op[3] = f2b(v.w * rstd * wv.w);
}

// ---------------- WMMA GEMM: C[M,N] = A[M,K] * W[N,K]^T + bias ----------------
// Block tile 128x128, 8 waves (4Mx2N), wave tile 32x64, KT=64, double-buffered LDS.
// EPI 0: fp32 out   EPI 1: bf16 to [B,H,S,HD] with scale   EPI 2: fp32 + resid
template <int EPI>
__global__ __launch_bounds__(256) void k_gemm(
    const unsigned short* __restrict__ A, const unsigned short* __restrict__ W,
    const float* __restrict__ bias, void* __restrict__ outp,
    const float* __restrict__ resid, float scale, int Mdim, int Ndim, int Kdim) {
  __shared__ unsigned short As[2][128][72];  // 64 K + 8 pad, rows 144B
  __shared__ unsigned short Wt[2][128][72];
  const int tid  = threadIdx.x;
  const int wave = tid >> 5, lane = tid & 31, l15 = lane & 15, kh = lane >> 4;
  const int m0 = blockIdx.y * 128, n0 = blockIdx.x * 128;
  const int wm = (wave >> 1) * 32, wn = (wave & 1) * 64;

  const int srow = tid >> 1, scol = (tid & 1) * 32;  // staging: 128 rows x 64 halfs
  const unsigned short* gA = A + (size_t)(m0 + srow) * Kdim + scol;
  const unsigned short* gW = W + (size_t)(n0 + srow) * Kdim + scol;

  v8f acc[2][4];
#pragma unroll
  for (int i = 0; i < 2; i++)
#pragma unroll
    for (int j = 0; j < 4; j++) acc[i][j] = (v8f){0, 0, 0, 0, 0, 0, 0, 0};

  auto stage = [&](int buf, int k0) {
#if USE_ASYNC_LDS
#pragma unroll
    for (int c = 0; c < 4; c++) {
      async_cp16(gA + k0 + c * 8, &As[buf][srow][scol + c * 8]);
      async_cp16(gW + k0 + c * 8, &Wt[buf][srow][scol + c * 8]);
    }
#else
#pragma unroll
    for (int c = 0; c < 4; c++) {
      *(uint4*)&As[buf][srow][scol + c * 8] = *(const uint4*)(gA + k0 + c * 8);
      *(uint4*)&Wt[buf][srow][scol + c * 8] = *(const uint4*)(gW + k0 + c * 8);
    }
#endif
  };

  stage(0, 0);
  int buf = 0;
  for (int k0 = 0; k0 < Kdim; k0 += 64) {
    async_wait0();
    __syncthreads();
    if (k0 + 64 < Kdim) stage(buf ^ 1, k0 + 64);  // DMA next tile over compute
#pragma unroll
    for (int ks = 0; ks < 2; ks++) {
      v16bf af[2];
#pragma unroll
      for (int mi = 0; mi < 2; mi++) {  // A frag: 16 contiguous K per lane
        const unsigned short* p = &As[buf][wm + mi * 16 + l15][ks * 32];
        af[mi] = cat16(*(const v8bf*)(p + kh * 16), *(const v8bf*)(p + kh * 16 + 8));
      }
#pragma unroll
      for (int ni = 0; ni < 4; ni++) {  // B frag: 16 contiguous K per lane
        const unsigned short* p = &Wt[buf][wn + ni * 16 + l15][ks * 32 + kh * 16];
        v16bf bfg = cat16(*(const v8bf*)p, *(const v8bf*)(p + 8));
#pragma unroll
        for (int mi = 0; mi < 2; mi++)
          acc[mi][ni] = __builtin_amdgcn_wmma_f32_16x16x32_bf16(
              false, af[mi], false, bfg, (short)0, acc[mi][ni], false, false);
      }
    }
    buf ^= 1;
  }

#pragma unroll
  for (int mi = 0; mi < 2; mi++)
#pragma unroll
    for (int ni = 0; ni < 4; ni++)
#pragma unroll
      for (int r = 0; r < 8; r++) {
        const int m = m0 + wm + mi * 16 + kh * 8 + r;
        const int n = n0 + wn + ni * 16 + l15;
        float v = acc[mi][ni][r] + bias[n];
        if (EPI == 0) {
          ((float*)outp)[(size_t)m * Ndim + n] = v;
        } else if (EPI == 2) {
          ((float*)outp)[(size_t)m * Ndim + n] = v + resid[(size_t)m * Ndim + n];
        } else {  // QKV repack: bf16 [B,H,S,HD]
          v *= scale;
          const int bb = m >> 11, ssi = m & (Sc - 1);
          const int hh = n >> 6,  hd  = n & (HDc - 1);
          ((unsigned short*)outp)[(((size_t)bb * Hc + hh) * Sc + ssi) * HDc + hd] = f2b(v);
        }
      }
}

// ---------------- flash attention (causal), WMMA bf16 ----------------
// Q already scaled by 1/sqrt(HD). Layout [B,H,S,HD] bf16. Output bf16 [B*S, D].
__global__ __launch_bounds__(256) void k_attn(
    const unsigned short* __restrict__ Q, const unsigned short* __restrict__ K,
    const unsigned short* __restrict__ V, unsigned short* __restrict__ O) {
  __shared__ unsigned short Ks[32][72];      // 32 keys x 64 d (+8 pad)
  __shared__ unsigned short Vt[64][40];      // V transposed: d x key
  __shared__ unsigned short Ps[8][16][40];   // per-wave P tile 16x32
  const int tid  = threadIdx.x;
  const int wave = tid >> 5, lane = tid & 31, l15 = lane & 15, kh = lane >> 4;
  const int b = blockIdx.z, h = blockIdx.y, q0 = blockIdx.x * 128;
  const int qbase = q0 + wave * 16;
  const size_t headoff = ((size_t)(b * Hc + h)) * Sc * HDc;
  const unsigned short* Qp = Q + headoff;
  const unsigned short* Kp = K + headoff;
  const unsigned short* Vp = V + headoff;

  v16bf qa[2];  // Q 16x64: two K=32 fragments, register-resident
#pragma unroll
  for (int kd = 0; kd < 2; kd++) {
    const unsigned short* qp = Qp + (size_t)(qbase + l15) * HDc + kd * 32;
    qa[kd] = cat16(*(const v8bf*)(qp + kh * 16), *(const v8bf*)(qp + kh * 16 + 8));
  }

  v8f oacc[4];
#pragma unroll
  for (int dt = 0; dt < 4; dt++) oacc[dt] = (v8f){0, 0, 0, 0, 0, 0, 0, 0};
  float mi_[8], li_[8];
#pragma unroll
  for (int r = 0; r < 8; r++) { mi_[r] = -1e30f; li_[r] = 0.f; }

  const int srow = tid >> 3, scol = (tid & 7) * 8;

  for (int kk0 = 0; kk0 < q0 + 128; kk0 += 32) {
    __syncthreads();
#if USE_ASYNC_LDS
    async_cp16(Kp + (size_t)(kk0 + srow) * HDc + scol, &Ks[srow][scol]);
#else
    *(uint4*)&Ks[srow][scol] =
        *(const uint4*)(Kp + (size_t)(kk0 + srow) * HDc + scol);
#endif
    uint4 vv = *(const uint4*)(Vp + (size_t)(kk0 + srow) * HDc + scol);
    const unsigned short* pv = (const unsigned short*)&vv;
#pragma unroll
    for (int i = 0; i < 8; i++) Vt[scol + i][srow] = pv[i];  // transpose V
    async_wait0();
    __syncthreads();

    // scores 16x32 = Q(16x64) @ K^T(64x32)
    v8f sc[2];
    sc[0] = (v8f){0, 0, 0, 0, 0, 0, 0, 0};
    sc[1] = (v8f){0, 0, 0, 0, 0, 0, 0, 0};
#pragma unroll
    for (int ni = 0; ni < 2; ni++)
#pragma unroll
      for (int kd = 0; kd < 2; kd++) {
        const unsigned short* kp = &Ks[ni * 16 + l15][kd * 32 + kh * 16];
        sc[ni] = __builtin_amdgcn_wmma_f32_16x16x32_bf16(
            false, qa[kd], false, cat16(*(const v8bf*)kp, *(const v8bf*)(kp + 8)),
            (short)0, sc[ni], false, false);
      }

    // causal mask + online softmax; DPP 16-lane row reductions (no LDS traffic)
#pragma unroll
    for (int r = 0; r < 8; r++) {
      const int qg = qbase + kh * 8 + r;
      float s0 = (kk0 + l15 <= qg) ? sc[0][r] : -1e30f;
      float s1 = (kk0 + 16 + l15 <= qg) ? sc[1][r] : -1e30f;
      const float t = redmax16(fmaxf(s0, s1));
      const float mnew = fmaxf(mi_[r], t);
      const float fs = __expf(mi_[r] - mnew);
      const float p0 = __expf(s0 - mnew);
      const float p1 = __expf(s1 - mnew);
      li_[r] = li_[r] * fs + redsum16(p0 + p1);
      mi_[r] = mnew;
#pragma unroll
      for (int dt = 0; dt < 4; dt++) oacc[dt][r] *= fs;
      Ps[wave][kh * 8 + r][l15]      = f2b(p0);  // C-layout -> LDS
      Ps[wave][kh * 8 + r][16 + l15] = f2b(p1);
    }
    asm volatile("" ::: "memory");  // keep in-wave DS store->load order

    // P A-fragment (16 contiguous K per lane), shared across the 4 d-tiles
    const unsigned short* pp = &Ps[wave][l15][0];
    v16bf pa = cat16(*(const v8bf*)(pp + kh * 16), *(const v8bf*)(pp + kh * 16 + 8));
#pragma unroll
    for (int dt = 0; dt < 4; dt++) {  // O += P(16x32) @ V(32x64)
      const unsigned short* vtp = &Vt[dt * 16 + l15][kh * 16];
      oacc[dt] = __builtin_amdgcn_wmma_f32_16x16x32_bf16(
          false, pa, false, cat16(*(const v8bf*)vtp, *(const v8bf*)(vtp + 8)),
          (short)0, oacc[dt], false, false);
    }
  }

#pragma unroll
  for (int dt = 0; dt < 4; dt++)
#pragma unroll
    for (int r = 0; r < 8; r++) {
      const int qg = qbase + kh * 8 + r;
      const int col = h * HDc + dt * 16 + l15;
      O[(size_t)(b * Sc + qg) * Dc + col] = f2b(oacc[dt][r] / li_[r]);
    }
}

// ---------------- silu(gate) * in -> bf16 ----------------
__global__ void k_silu_mul(const float* __restrict__ g,
                           const float* __restrict__ v,
                           unsigned short* __restrict__ o, size_t n) {
  for (size_t i = (size_t)blockIdx.x * blockDim.x + threadIdx.x; i < n;
       i += (size_t)gridDim.x * blockDim.x) {
    const float gg = g[i];
    const float si = gg / (1.0f + __expf(-gg));
    o[i] = f2b(si * v[i]);
  }
}

// ---------------- launch ----------------
extern "C" void kernel_launch(void* const* d_in, const int* in_sizes, int n_in,
                              void* d_out, int out_size, void* d_ws, size_t ws_size,
                              hipStream_t stream) {
  const float* x    = (const float*)d_in[0];
  const float* pos  = (const float*)d_in[1];
  const float* n1w  = (const float*)d_in[2];
  const float* n2w  = (const float*)d_in[3];
  const float* qW   = (const float*)d_in[4];  const float* qB   = (const float*)d_in[5];
  const float* kW   = (const float*)d_in[6];  const float* kB   = (const float*)d_in[7];
  const float* vW   = (const float*)d_in[8];  const float* vB   = (const float*)d_in[9];
  const float* oW   = (const float*)d_in[10]; const float* oB   = (const float*)d_in[11];
  const float* inW  = (const float*)d_in[12]; const float* inB  = (const float*)d_in[13];
  const float* gW   = (const float*)d_in[14]; const float* gB   = (const float*)d_in[15];
  const float* outW = (const float*)d_in[16]; const float* outB = (const float*)d_in[17];

  char* ws = (char*)d_ws;
  size_t off = 0;
  auto alloc = [&](size_t bytes) -> void* {
    void* p = ws + off;
    off = (off + bytes + 255) & ~(size_t)255;
    return p;
  };

  unsigned short* qWb   = (unsigned short*)alloc((size_t)Dc * Dc * 2);
  unsigned short* kWb   = (unsigned short*)alloc((size_t)Dc * Dc * 2);
  unsigned short* vWb   = (unsigned short*)alloc((size_t)Dc * Dc * 2);
  unsigned short* oWb   = (unsigned short*)alloc((size_t)Dc * Dc * 2);
  unsigned short* inWb  = (unsigned short*)alloc((size_t)HIDc * Dc * 2);
  unsigned short* gWb   = (unsigned short*)alloc((size_t)HIDc * Dc * 2);
  unsigned short* outWb = (unsigned short*)alloc((size_t)Dc * HIDc * 2);
  float*          xr    = (float*)alloc((size_t)Mc * Dc * 4);
  float*          xr2   = (float*)alloc((size_t)Mc * Dc * 4);
  unsigned short* hbf   = (unsigned short*)alloc((size_t)Mc * Dc * 2);
  unsigned short* qb    = (unsigned short*)alloc((size_t)Mc * Dc * 2);
  unsigned short* kb    = (unsigned short*)alloc((size_t)Mc * Dc * 2);
  unsigned short* vb    = (unsigned short*)alloc((size_t)Mc * Dc * 2);
  unsigned short* attnb = (unsigned short*)alloc((size_t)Mc * Dc * 2);
  float*          gatef = (float*)alloc((size_t)Mc * HIDc * 4);
  float*          inf_  = (float*)alloc((size_t)Mc * HIDc * 4);
  unsigned short* hmid  = (unsigned short*)alloc((size_t)Mc * HIDc * 2);

  // 1) weights -> bf16
  struct { const float* s; unsigned short* d; size_t n; } cv[7] = {
      {qW, qWb, (size_t)Dc * Dc},    {kW, kWb, (size_t)Dc * Dc},
      {vW, vWb, (size_t)Dc * Dc},    {oW, oWb, (size_t)Dc * Dc},
      {inW, inWb, (size_t)HIDc * Dc},{gW, gWb, (size_t)HIDc * Dc},
      {outW, outWb, (size_t)Dc * HIDc}};
  for (int i = 0; i < 7; i++) {
    int blocks = (int)((cv[i].n + 1023) / 1024);
    k_f32_to_bf16<<<blocks, 1024, 0, stream>>>(cv[i].s, cv[i].d, cv[i].n);
  }

  // 2) xr = x + pos; hbf = bf16(rmsnorm(xr) * norm1_w)
  k_rmsnorm<true><<<Mc, 256, 0, stream>>>(x, pos, n1w, xr, hbf);

  // 3) QKV projections -> bf16 [B,H,S,HD]; Q pre-scaled by 1/sqrt(HD)
  dim3 gqkv(Dc / 128, Mc / 128);
  k_gemm<1><<<gqkv, 256, 0, stream>>>(hbf, qWb, qB, qb, nullptr, 0.125f, Mc, Dc, Dc);
  k_gemm<1><<<gqkv, 256, 0, stream>>>(hbf, kWb, kB, kb, nullptr, 1.0f,   Mc, Dc, Dc);
  k_gemm<1><<<gqkv, 256, 0, stream>>>(hbf, vWb, vB, vb, nullptr, 1.0f,   Mc, Dc, Dc);

  // 4) causal flash attention -> bf16 [M, D]
  dim3 ga(Sc / 128, Hc, Bc);
  k_attn<<<ga, 256, 0, stream>>>(qb, kb, vb, attnb);

  // 5) O projection + residual -> xr2 (fp32)
  k_gemm<2><<<gqkv, 256, 0, stream>>>(attnb, oWb, oB, xr2, xr, 1.0f, Mc, Dc, Dc);

  // 6) second RMSNorm -> hbf (reuse)
  k_rmsnorm<false><<<Mc, 256, 0, stream>>>(xr2, nullptr, n2w, nullptr, hbf);

  // 7) MLP gate / in GEMMs (fp32 out)
  dim3 gmlp(HIDc / 128, Mc / 128);
  k_gemm<0><<<gmlp, 256, 0, stream>>>(hbf, gWb,  gB,  gatef, nullptr, 1.0f, Mc, HIDc, Dc);
  k_gemm<0><<<gmlp, 256, 0, stream>>>(hbf, inWb, inB, inf_,  nullptr, 1.0f, Mc, HIDc, Dc);

  // 8) h = silu(gate) * in -> bf16
  k_silu_mul<<<4096, 256, 0, stream>>>(gatef, inf_, hmid, (size_t)Mc * HIDc);

  // 9) out projection + residual -> d_out (fp32)
  k_gemm<2><<<gqkv, 256, 0, stream>>>(hmid, outWb, outB, (float*)d_out, xr2, 1.0f,
                                      Mc, Dc, HIDc);
}
